// Seq2Seq_88235808129357
// MI455X (gfx1250) — matrix operand order repaired and checked by reading, not compiled, not run on hardware
//
#include <hip/hip_runtime.h>
#include <hip/hip_bf16.h>
#include <stdint.h>

// ---------------------------------------------------------------------------
// Seq2Seq LSTM (S=128,B=64,I=H=O=1024,L=2,T=64) for gfx1250 (MI455X).
// bf16 weights resident in L2 (192MB), v_wmma_f32_16x16x32_bf16 matrix cores,
// one launch per recurrent cell (stream-ordered recurrence, graph-replayed).
// ---------------------------------------------------------------------------

typedef __attribute__((ext_vector_type(16))) __bf16 v16bf;
typedef __attribute__((ext_vector_type(8)))  __bf16 v8bf;
typedef __attribute__((ext_vector_type(8)))  float  v8f;

#define SQ_S 128
#define SQ_B 64
#define SQ_H 1024
#define SQ_G 4096   // 4*H
#define SQ_O 1024
#define SQ_T 64
#define SQ_L 2

// ---------------------------------------------------------------------------
// fp32 -> bf16 conversion (vectorized by 4), grid-stride
// ---------------------------------------------------------------------------
__global__ void cvt_f32_bf16(const float* __restrict__ s, __bf16* __restrict__ d, int n4) {
    int i = blockIdx.x * blockDim.x + threadIdx.x;
    int stride = gridDim.x * blockDim.x;
    for (; i < n4; i += stride) {
        float4 v = ((const float4*)s)[i];
        d[4 * i + 0] = (__bf16)v.x;
        d[4 * i + 1] = (__bf16)v.y;
        d[4 * i + 2] = (__bf16)v.z;
        d[4 * i + 3] = (__bf16)v.w;
    }
}

__global__ void zero_u32(uint32_t* __restrict__ p, int n) {
    int i = blockIdx.x * blockDim.x + threadIdx.x;
    if (i < n) p[i] = 0u;
}

__device__ __forceinline__ float sigmoidf_(float x) {
    return 1.0f / (1.0f + __expf(-x));
}

// ---------------------------------------------------------------------------
// One LSTM cell step:  g = x@wih.T + h@whh.T (+bias in epilogue)
//   x, h_in : [64,1024] bf16       wih, whh : [4096,1024] bf16 (row-major N,K)
//   bias    : [4096] fp32          c : [64,1024] fp32 (in/out, owned per-WG)
//   h_out   : [64,1024] bf16       h_out2 : optional duplicate (enc layer-0 ys)
// Grid: 64 workgroups, each owns 16 columns j0..j0+15 across all 4 gates.
// Block: 256 threads = 8 waves; wave (gate = w>>1, mhalf = w&1) computes
// two 16x16 M-tiles of its gate via v_wmma_f32_16x16x32_bf16, K-loop 2*1024.
// ---------------------------------------------------------------------------
__global__ __launch_bounds__(256)
void lstm_cell_kernel(const __bf16* __restrict__ x,
                      const __bf16* __restrict__ h_in,
                      const __bf16* __restrict__ wih,
                      const __bf16* __restrict__ whh,
                      const float*  __restrict__ bias,
                      float*        __restrict__ c,
                      __bf16*       __restrict__ h_out,
                      __bf16*       __restrict__ h_out2) {
    __shared__ float gsm[4][SQ_B][16];   // 16 KB pre-activation staging

    const int tid   = threadIdx.x;
    const int wave  = tid >> 5;
    const int lane  = tid & 31;
    const int gate  = wave >> 1;         // 0..3 : i,f,g,o
    const int mhalf = wave & 1;          // M-tiles {2*mhalf, 2*mhalf+1}
    const int j0    = blockIdx.x * 16;   // column slice within H
    const int l16   = lane & 15;
    const int lhi   = lane >> 4;         // 0: lanes 0-15, 1: lanes 16-31

    // bf16 A-matrix 16x32 striping: lanes 0-15 elems = K{0..7,16..23},
    // lanes 16-31 elems = K{8..15,24..31}
    const int aklo = lhi * 8;
    // bf16 B-matrix 32x16 striping: lanes 0-15 K{0..7,16..23}, lanes 16-31 K{8..15,24..31}
    const int bklo = lhi * 8;

    const int am0 = (mhalf * 2 + 0) * 16 + l16;   // batch rows for A frags
    const int am1 = (mhalf * 2 + 1) * 16 + l16;
    const int nrow = gate * SQ_H + j0 + l16;      // weight row (N index)

    v8f acc0 = {};
    v8f acc1 = {};

    #pragma unroll
    for (int pass = 0; pass < 2; ++pass) {
        const __bf16* A = pass ? h_in : x;
        const __bf16* W = pass ? whh : wih;
        const __bf16* wrow = W + (size_t)nrow * SQ_H;
        const __bf16* arow0 = A + (size_t)am0 * SQ_H;
        const __bf16* arow1 = A + (size_t)am1 * SQ_H;

        #pragma unroll 2
        for (int k0 = 0; k0 < SQ_H; k0 += 32) {
            __builtin_prefetch(wrow + k0 + 128, 0, 0);  // global_prefetch_b8

            v8bf blo = *(const v8bf*)(wrow + k0 + bklo);
            v8bf bhi = *(const v8bf*)(wrow + k0 + bklo + 16);
            v16bf bf = __builtin_shufflevector(blo, bhi,
                0, 1, 2, 3, 4, 5, 6, 7, 8, 9, 10, 11, 12, 13, 14, 15);

            v8bf a0l = *(const v8bf*)(arow0 + k0 + aklo);
            v8bf a0h = *(const v8bf*)(arow0 + k0 + aklo + 16);
            v16bf a0 = __builtin_shufflevector(a0l, a0h,
                0, 1, 2, 3, 4, 5, 6, 7, 8, 9, 10, 11, 12, 13, 14, 15);

            v8bf a1l = *(const v8bf*)(arow1 + k0 + aklo);
            v8bf a1h = *(const v8bf*)(arow1 + k0 + aklo + 16);
            v16bf a1 = __builtin_shufflevector(a1l, a1h,
                0, 1, 2, 3, 4, 5, 6, 7, 8, 9, 10, 11, 12, 13, 14, 15);

            acc0 = __builtin_amdgcn_wmma_f32_16x16x32_bf16(
                false, a0, false, bf, (short)0, acc0, false, false);
            acc1 = __builtin_amdgcn_wmma_f32_16x16x32_bf16(
                false, a1, false, bf, (short)0, acc1, false, false);
        }
    }

    // D layout: lanes 0-15 -> (M=r, N=lane); lanes 16-31 -> (M=r+8, N=lane-16)
    #pragma unroll
    for (int r = 0; r < 8; ++r) {
        int rr = lhi ? (r + 8) : r;
        gsm[gate][(mhalf * 2 + 0) * 16 + rr][l16] = acc0[r];
        gsm[gate][(mhalf * 2 + 1) * 16 + rr][l16] = acc1[r];
    }
    __syncthreads();

    // gate nonlinearity epilogue: 64x16 elems, 4 per thread
    for (int e = tid; e < SQ_B * 16; e += 256) {
        int b  = e >> 4;
        int jj = e & 15;
        int j  = j0 + jj;
        float gi = gsm[0][b][jj] + bias[0 * SQ_H + j];
        float gf = gsm[1][b][jj] + bias[1 * SQ_H + j];
        float gg = gsm[2][b][jj] + bias[2 * SQ_H + j];
        float go = gsm[3][b][jj] + bias[3 * SQ_H + j];
        float cf = c[b * SQ_H + j];
        float cn = sigmoidf_(gf) * cf + sigmoidf_(gi) * tanhf(gg);
        float hn = sigmoidf_(go) * tanhf(cn);
        c[b * SQ_H + j] = cn;
        __bf16 hb = (__bf16)hn;
        h_out[b * SQ_H + j] = hb;
        if (h_out2) h_out2[b * SQ_H + j] = hb;
    }
}

// ---------------------------------------------------------------------------
// Decoder head: y = h @ lin_w.T + lin_b
//   out fp32 strided into d_out[b][t][:], x_next bf16 feedback buffer.
// Grid 32 WGs x 32 cols; 8 waves, each one 16x16 tile (mt = w&3, nt = w>>2).
// ---------------------------------------------------------------------------
__global__ __launch_bounds__(256)
void linear_kernel(const __bf16* __restrict__ h,
                   const __bf16* __restrict__ w,
                   const float*  __restrict__ bias,
                   float*        __restrict__ out,
                   int out_row_stride,
                   __bf16*       __restrict__ x_next) {
    const int tid  = threadIdx.x;
    const int wave = tid >> 5;
    const int lane = tid & 31;
    const int mt   = wave & 3;
    const int nt   = wave >> 2;
    const int l16  = lane & 15;
    const int lhi  = lane >> 4;
    const int aklo = lhi * 8;
    const int bklo = lhi * 8;

    const int n0   = blockIdx.x * 32 + nt * 16;
    const int am   = mt * 16 + l16;
    const int nrow = n0 + l16;

    const __bf16* wrow = w + (size_t)nrow * SQ_H;
    const __bf16* arow = h + (size_t)am * SQ_H;

    v8f acc = {};
    #pragma unroll 2
    for (int k0 = 0; k0 < SQ_H; k0 += 32) {
        v8bf blo = *(const v8bf*)(wrow + k0 + bklo);
        v8bf bhi = *(const v8bf*)(wrow + k0 + bklo + 16);
        v16bf bf = __builtin_shufflevector(blo, bhi,
            0, 1, 2, 3, 4, 5, 6, 7, 8, 9, 10, 11, 12, 13, 14, 15);
        v8bf al = *(const v8bf*)(arow + k0 + aklo);
        v8bf ah = *(const v8bf*)(arow + k0 + aklo + 16);
        v16bf af = __builtin_shufflevector(al, ah,
            0, 1, 2, 3, 4, 5, 6, 7, 8, 9, 10, 11, 12, 13, 14, 15);
        acc = __builtin_amdgcn_wmma_f32_16x16x32_bf16(
            false, af, false, bf, (short)0, acc, false, false);
    }

    const int colg = n0 + l16;
    const float bv = bias[colg];
    #pragma unroll
    for (int r = 0; r < 8; ++r) {
        int row = mt * 16 + (lhi ? (r + 8) : r);
        float v = acc[r] + bv;
        out[(size_t)row * out_row_stride + colg] = v;
        x_next[row * SQ_H + colg] = (__bf16)v;
    }
}

// ---------------------------------------------------------------------------
// Host orchestration
// ---------------------------------------------------------------------------
static inline void launch_cvt(const float* s, __bf16* d, size_t n, hipStream_t st) {
    int n4 = (int)(n / 4);
    int blocks = (n4 + 255) / 256;
    if (blocks > 4096) blocks = 4096;
    cvt_f32_bf16<<<blocks, 256, 0, st>>>(s, d, n4);
}

extern "C" void kernel_launch(void* const* d_in, const int* in_sizes, int n_in,
                              void* d_out, int out_size, void* d_ws, size_t ws_size,
                              hipStream_t stream) {
    const float* input_seq = (const float*)d_in[0];   // [S,B,I]
    const float* enc_wih_f = (const float*)d_in[1];   // [L,4H,I]
    const float* enc_whh_f = (const float*)d_in[2];   // [L,4H,H]
    const float* enc_b     = (const float*)d_in[3];   // [L,4H]
    const float* dec_wih_f = (const float*)d_in[4];
    const float* dec_whh_f = (const float*)d_in[5];
    const float* dec_b     = (const float*)d_in[6];
    const float* lin_w_f   = (const float*)d_in[7];   // [O,H]
    const float* lin_b     = (const float*)d_in[8];   // [O]
    float* out = (float*)d_out;                       // [B,T,O]

    (void)in_sizes; (void)n_in; (void)out_size; (void)ws_size;

    // ---- workspace carve-up (~100 MB) ----
    uint8_t* ws = (uint8_t*)d_ws;
    size_t off = 0;
    auto carve = [&](size_t bytes) -> void* {
        void* p = ws + off;
        off += (bytes + 255) & ~(size_t)255;
        return p;
    };
    const size_t WN  = (size_t)SQ_L * SQ_G * SQ_H;  // per weight tensor elems
    const size_t XN  = (size_t)SQ_S * SQ_B * SQ_H;  // input seq elems
    const size_t HN  = (size_t)SQ_B * SQ_H;

    __bf16* encWih = (__bf16*)carve(WN * 2);
    __bf16* encWhh = (__bf16*)carve(WN * 2);
    __bf16* decWih = (__bf16*)carve(WN * 2);
    __bf16* decWhh = (__bf16*)carve(WN * 2);
    __bf16* linW   = (__bf16*)carve((size_t)SQ_O * SQ_H * 2);
    __bf16* xseq   = (__bf16*)carve(XN * 2);
    __bf16* ys0    = (__bf16*)carve(XN * 2);        // enc layer-0 outputs
    __bf16* h0p[2] = { (__bf16*)carve(HN * 2), (__bf16*)carve(HN * 2) };
    __bf16* h1p[2] = { (__bf16*)carve(HN * 2), (__bf16*)carve(HN * 2) };
    __bf16* xbuf   = (__bf16*)carve(HN * 2);        // decoder feedback input
    float*  c0     = (float*)carve(HN * 4);
    float*  c1     = (float*)carve(HN * 4);

    // ---- setup: bf16 conversions + zero initial state (every call) ----
    launch_cvt(enc_wih_f, encWih, WN, stream);
    launch_cvt(enc_whh_f, encWhh, WN, stream);
    launch_cvt(dec_wih_f, decWih, WN, stream);
    launch_cvt(dec_whh_f, decWhh, WN, stream);
    launch_cvt(lin_w_f,   linW,  (size_t)SQ_O * SQ_H, stream);
    launch_cvt(input_seq, xseq,  XN, stream);

    {
        int nh = (int)(HN * 2 / 4);   // bf16 buffer in u32 words
        int nc = (int)(HN * 4 / 4);   // f32 buffer in u32 words
        zero_u32<<<(nh + 255) / 256, 256, 0, stream>>>((uint32_t*)h0p[0], nh);
        zero_u32<<<(nh + 255) / 256, 256, 0, stream>>>((uint32_t*)h1p[0], nh);
        zero_u32<<<(nc + 255) / 256, 256, 0, stream>>>((uint32_t*)c0, nc);
        zero_u32<<<(nc + 255) / 256, 256, 0, stream>>>((uint32_t*)c1, nc);
    }

    const size_t LW = (size_t)SQ_G * SQ_H;   // per-layer weight elems
    dim3 cgrid(64), cblk(256);

    // ---- encoder layer 0 over full sequence (spill ys0) ----
    int cur0 = 0;
    for (int t = 0; t < SQ_S; ++t) {
        lstm_cell_kernel<<<cgrid, cblk, 0, stream>>>(
            xseq + (size_t)t * HN, h0p[cur0],
            encWih, encWhh, enc_b,
            c0, h0p[cur0 ^ 1], ys0 + (size_t)t * HN);
        cur0 ^= 1;
    }
    // ---- encoder layer 1 over ys0 ----
    int cur1 = 0;
    for (int t = 0; t < SQ_S; ++t) {
        lstm_cell_kernel<<<cgrid, cblk, 0, stream>>>(
            ys0 + (size_t)t * HN, h1p[cur1],
            encWih + LW, encWhh + LW, enc_b + SQ_G,
            c1, h1p[cur1 ^ 1], nullptr);
        cur1 ^= 1;
    }

    // ---- autoregressive decoder (states carry over from encoder) ----
    for (int t = 0; t < SQ_T; ++t) {
        const __bf16* x = (t == 0) ? (xseq + (size_t)(SQ_S - 1) * HN) : xbuf;
        lstm_cell_kernel<<<cgrid, cblk, 0, stream>>>(
            x, h0p[cur0],
            decWih, decWhh, dec_b,
            c0, h0p[cur0 ^ 1], nullptr);
        lstm_cell_kernel<<<cgrid, cblk, 0, stream>>>(
            h0p[cur0 ^ 1], h1p[cur1],
            decWih + LW, decWhh + LW, dec_b + SQ_G,
            c1, h1p[cur1 ^ 1], nullptr);
        linear_kernel<<<dim3(32), cblk, 0, stream>>>(
            h1p[cur1 ^ 1], linW, lin_b,
            out + (size_t)t * SQ_O, SQ_T * SQ_O, xbuf);
        cur0 ^= 1;
        cur1 ^= 1;
    }
}
